// ExpertGatedAggregator_86406152061591
// MI455X (gfx1250) — compile-verified, matrix operand order checked by reading, not run on hardware
//
#include <hip/hip_runtime.h>

// ---------------------------------------------------------------------------
// ExpertGatedAggregator for MI455X (gfx1250, wave32, WMMA)
// bf16 activations/weights, f32 accumulation via v_wmma_f32_16x16x32_bf16
// ---------------------------------------------------------------------------

typedef __bf16 bf16_t;
typedef bf16_t v16bf __attribute__((ext_vector_type(16)));
typedef float  v8f   __attribute__((ext_vector_type(8)));
typedef unsigned int u32x4 __attribute__((ext_vector_type(4)));

union FragU {
    v16bf v;     // 16 bf16 = 32 bytes = 8 VGPRs (one WMMA A/B operand per lane)
    u32x4 q[2];  // two 16-byte chunks
};

constexpr int E_   = 8;
constexpr int B_   = 32768;
constexpr int D_   = 256;
constexpr int H_   = 1024;
constexpr int GIN  = E_ * D_ + E_;   // 2056
constexpr int GINP = 2080;           // padded to multiple of 32 (K per WMMA)

// ---------------------------------------------------------------------------
// Pack gate_in = [reprs_cat | probs_cat | zero pad] as bf16 [B, 2080]
// ---------------------------------------------------------------------------
__global__ __launch_bounds__(256) void pack_gatein(
    const float* __restrict__ reprs,   // [E, B, D]
    const float* __restrict__ probs,   // [E, B, 1]
    bf16_t* __restrict__ out)          // [B, GINP]
{
    long idx = (long)blockIdx.x * blockDim.x + threadIdx.x;
    if (idx >= (long)B_ * GINP) return;
    int b   = (int)(idx / GINP);
    int col = (int)(idx % GINP);
    float v = 0.0f;
    if (col < E_ * D_) {
        int e = col >> 8;          // D_ == 256
        int d = col & (D_ - 1);
        v = reprs[((size_t)e * B_ + b) * D_ + d];
    } else if (col < GIN) {
        int e = col - E_ * D_;
        v = probs[(size_t)e * B_ + b];
    }
    out[idx] = (bf16_t)v;
}

// ---------------------------------------------------------------------------
// Weight transposes to bf16, K-major rows (row = output col n, cols = k)
// ---------------------------------------------------------------------------
__global__ __launch_bounds__(256) void transpose_w1(
    const float* __restrict__ W1,      // [GIN, H]
    bf16_t* __restrict__ W1t)          // [H, GINP]
{
    int idx = blockIdx.x * blockDim.x + threadIdx.x;
    if (idx >= H_ * GINP) return;
    int n = idx / GINP;
    int k = idx % GINP;
    float v = (k < GIN) ? W1[(size_t)k * H_ + n] : 0.0f;
    W1t[idx] = (bf16_t)v;
}

__global__ __launch_bounds__(256) void transpose_w2(
    const float* __restrict__ W2,      // [H, H/2]
    bf16_t* __restrict__ W2t)          // [H/2, H]
{
    int idx = blockIdx.x * blockDim.x + threadIdx.x;
    if (idx >= (H_ / 2) * H_) return;
    int n = idx / H_;
    int k = idx % H_;
    W2t[idx] = (bf16_t)W2[(size_t)k * (H_ / 2) + n];
}

// ---------------------------------------------------------------------------
// WMMA GEMM: C[M,N] = relu(A[M,K] @ Bt[N,K]^T + bias[N])
// One wave -> 16x(16*NT) output strip. A fragment reused across NT WMMAs.
// A-fragment layout (16-bit A 16x32, ISA 7.12.2):
//   lane l: M = l&15, hk = l>>4; V0..3 = K 8*hk+0..7, V4..7 = K 16+8*hk+0..7
// Bt rows are output columns, so B fragments load with the identical pattern.
// C/D: lane l holds column l&15; VGPR r holds row 8*(l>>4)+r.
// ---------------------------------------------------------------------------
template <bool BF16OUT>
__global__ __launch_bounds__(256) void gemm_wmma(
    const bf16_t* __restrict__ A,  int lda,
    const bf16_t* __restrict__ Bt, int ldb,
    const float*  __restrict__ bias,
    bf16_t* __restrict__ outBf, float* __restrict__ outF, int ldc,
    int ksteps, int ngroups)
{
    constexpr int NT = 8;
    int wave = (int)((blockIdx.x * blockDim.x + threadIdx.x) >> 5);
    int lane = threadIdx.x & 31;
    int l15  = lane & 15;
    int hk   = lane >> 4;

    int ngroup = wave % ngroups;
    int mtile  = wave / ngroups;

    const bf16_t* arow = A  + (size_t)(mtile * 16 + l15) * lda + 8 * hk;
    const bf16_t* brow = Bt + (size_t)(ngroup * (NT * 16) + l15) * ldb + 8 * hk;

    v8f acc[NT] = {};

    for (int ks = 0; ks < ksteps; ++ks) {
        const int k0 = ks * 32;
        FragU a;
        a.q[0] = *(const u32x4*)(arow + k0);
        a.q[1] = *(const u32x4*)(arow + k0 + 16);
#pragma unroll
        for (int t = 0; t < NT; ++t) {
            FragU b;
            const bf16_t* bp = brow + (size_t)(t * 16) * ldb + k0;
            b.q[0] = *(const u32x4*)(bp);
            b.q[1] = *(const u32x4*)(bp + 16);
            acc[t] = __builtin_amdgcn_wmma_f32_16x16x32_bf16(
                /*neg_a=*/false, a.v, /*neg_b=*/false, b.v,
                /*c_mod=*/(short)0, acc[t],
                /*reuse_a=*/false, /*reuse_b=*/false);
        }
    }

    const int m0 = mtile * 16 + hk * 8;
#pragma unroll
    for (int t = 0; t < NT; ++t) {
        int   n  = ngroup * (NT * 16) + t * 16 + l15;
        float bv = bias[n];
#pragma unroll
        for (int r = 0; r < 8; ++r) {
            float v = acc[t][r] + bv;
            v = v > 0.0f ? v : 0.0f;
            size_t o = (size_t)(m0 + r) * ldc + n;
            if (BF16OUT) outBf[o] = (bf16_t)v;
            else         outF[o]  = v;
        }
    }
}

// ---------------------------------------------------------------------------
// Finalize: logits = h2 @ W3 + b3; softmax; *mask; renorm; weighted prob.
// One thread per row; W3 staged in LDS (16 KB).
// d_out layout: [0,B) weighted_prob, [B, B+8B) gate_w row-major.
// ---------------------------------------------------------------------------
__global__ __launch_bounds__(256) void gate_finalize(
    const float* __restrict__ h2,     // [B, 512]
    const float* __restrict__ W3,     // [512, 8]
    const float* __restrict__ b3,     // [8]
    const float* __restrict__ probs,  // [E, B, 1]
    const float* __restrict__ mask,   // [B, E]
    float* __restrict__ out)
{
    __shared__ float w3s[(H_ / 2) * E_];
    for (int i = threadIdx.x; i < (H_ / 2) * E_; i += blockDim.x) w3s[i] = W3[i];
    __syncthreads();

    int b = blockIdx.x * blockDim.x + threadIdx.x;
    if (b >= B_) return;

    float logit[E_];
#pragma unroll
    for (int e = 0; e < E_; ++e) logit[e] = b3[e];

    const float4* h2r = (const float4*)(h2 + (size_t)b * (H_ / 2));
    for (int k4 = 0; k4 < (H_ / 2) / 4; ++k4) {
        float4 hv = h2r[k4];
        const float* wp = &w3s[k4 * 4 * E_];
#pragma unroll
        for (int e = 0; e < E_; ++e) {
            logit[e] += hv.x * wp[e] + hv.y * wp[E_ + e] +
                        hv.z * wp[2 * E_ + e] + hv.w * wp[3 * E_ + e];
        }
    }

    float mx = logit[0];
#pragma unroll
    for (int e = 1; e < E_; ++e) mx = logit[e] > mx ? logit[e] : mx;

    float gw[E_];
    float s = 0.0f;
#pragma unroll
    for (int e = 0; e < E_; ++e) { gw[e] = __expf(logit[e] - mx); s += gw[e]; }

    float s2 = 0.0f;
    float inv_s = 1.0f / s;
#pragma unroll
    for (int e = 0; e < E_; ++e) {
        gw[e] = gw[e] * inv_s * mask[(size_t)b * E_ + e];
        s2 += gw[e];
    }

    float inv = 1.0f / (s2 + 1e-8f);
    float wprob = 0.0f;
#pragma unroll
    for (int e = 0; e < E_; ++e) {
        gw[e] *= inv;
        wprob += gw[e] * probs[(size_t)e * B_ + b];
    }

    out[b] = wprob;
#pragma unroll
    for (int e = 0; e < E_; ++e) out[(size_t)B_ + (size_t)b * E_ + e] = gw[e];
}

// ---------------------------------------------------------------------------
// Launch
// ---------------------------------------------------------------------------
extern "C" void kernel_launch(void* const* d_in, const int* in_sizes, int n_in,
                              void* d_out, int out_size, void* d_ws, size_t ws_size,
                              hipStream_t stream)
{
    (void)in_sizes; (void)n_in; (void)out_size; (void)ws_size;

    const float* reprs = (const float*)d_in[0];  // [E,B,D]
    const float* probs = (const float*)d_in[1];  // [E,B,1]
    const float* mask  = (const float*)d_in[2];  // [B,E]
    const float* W1    = (const float*)d_in[3];  // [GIN,H]
    const float* b1    = (const float*)d_in[4];  // [H]
    const float* W2    = (const float*)d_in[5];  // [H,H/2]
    const float* b2    = (const float*)d_in[6];  // [H/2]
    const float* W3    = (const float*)d_in[7];  // [H/2,E]
    const float* b3    = (const float*)d_in[8];  // [E]
    float* out = (float*)d_out;

    char* ws = (char*)d_ws;
    size_t off = 0;
    auto take = [&](size_t bytes) -> void* {
        void* p = ws + off;
        off = (off + bytes + 255) & ~(size_t)255;
        return p;
    };
    bf16_t* gate_in = (bf16_t*)take((size_t)B_ * GINP * sizeof(bf16_t));       // 136 MB
    bf16_t* W1t     = (bf16_t*)take((size_t)H_ * GINP * sizeof(bf16_t));       // 4.3 MB
    bf16_t* W2t     = (bf16_t*)take((size_t)(H_ / 2) * H_ * sizeof(bf16_t));   // 1 MB
    bf16_t* h1      = (bf16_t*)take((size_t)B_ * H_ * sizeof(bf16_t));         // 64 MB
    float*  h2      = (float*) take((size_t)B_ * (H_ / 2) * sizeof(float));    // 64 MB

    {
        long total = (long)B_ * GINP;
        pack_gatein<<<(int)((total + 255) / 256), 256, 0, stream>>>(reprs, probs, gate_in);
    }
    transpose_w1<<<(H_ * GINP + 255) / 256, 256, 0, stream>>>(W1, W1t);
    transpose_w2<<<((H_ / 2) * H_ + 255) / 256, 256, 0, stream>>>(W2, W2t);

    // GEMM1: M=32768 (2048 M-tiles), N=1024 (8 n-groups of 128), K=2080 (65 steps)
    {
        int mtiles = B_ / 16, ngroups = H_ / 128;
        int waves  = mtiles * ngroups;                 // 16384
        gemm_wmma<true><<<waves * 32 / 256, 256, 0, stream>>>(
            gate_in, GINP, W1t, GINP, b1, h1, nullptr, H_, GINP / 32, ngroups);
    }
    // GEMM2: M=32768, N=512 (4 n-groups), K=1024 (32 steps)
    {
        int mtiles = B_ / 16, ngroups = (H_ / 2) / 128;
        int waves  = mtiles * ngroups;                 // 8192
        gemm_wmma<false><<<waves * 32 / 256, 256, 0, stream>>>(
            h1, H_, W2t, H_, b2, nullptr, h2, H_ / 2, H_ / 32, ngroups);
    }

    gate_finalize<<<B_ / 256, 256, 0, stream>>>(h2, W3, b3, probs, mask, out);
}
